// GCNConv_23295902613545
// MI455X (gfx1250) — matrix operand order repaired and checked by reading, not compiled, not run on hardware
//
#include <hip/hip_runtime.h>
#include <hip/hip_bf16.h>

#define N_NODES 50000
#define N_EDGES 300000
#define DFEAT   256

typedef __bf16 bf16_t;
typedef bf16_t v16bf __attribute__((ext_vector_type(16)));
typedef bf16_t v8bf  __attribute__((ext_vector_type(8)));
typedef bf16_t v4bf  __attribute__((ext_vector_type(4)));
typedef float  v8f   __attribute__((ext_vector_type(8)));
typedef float  v4f   __attribute__((ext_vector_type(4)));

// ---------------- zero the fp32 accumulator (float4 per thread) ----------------
__global__ __launch_bounds__(256) void zero_f32_k(float* __restrict__ p, int n4) {
    int i = blockIdx.x * 256 + threadIdx.x;
    if (i < n4) {
        v4f z = {0.f, 0.f, 0.f, 0.f};
        ((v4f*)p)[i] = z;
    }
}

// ---------------- scatter: agg[dst] += w_e * x[src]  (fp32 atomics) ----------------
// one thread per (edge, 4-feature chunk): 300000 * 64 threads
__global__ __launch_bounds__(256) void scatter_k(const float* __restrict__ x,
                                                 const int*   __restrict__ ei,
                                                 const float* __restrict__ ew,
                                                 float*       __restrict__ agg) {
    int idx = blockIdx.x * 256 + threadIdx.x;
    if (idx >= N_EDGES * 64) return;
    int e = idx >> 6;        // edge id
    int q = idx & 63;        // float4 chunk within 256 features
    int dst = ei[e];              // edge_index[0][e]
    int src = ei[N_EDGES + e];    // edge_index[1][e]
    float w = ew[e];
    v4f v = ((const v4f*)x)[(size_t)src * 64 + q];
    float* o = agg + (size_t)dst * DFEAT + q * 4;
    unsafeAtomicAdd(o + 0, w * v[0]);   // global_atomic_add_f32, no return
    unsafeAtomicAdd(o + 1, w * v[1]);
    unsafeAtomicAdd(o + 2, w * v[2]);
    unsafeAtomicAdd(o + 3, w * v[3]);
}

// ---------------- fp32 -> bf16 conversion (4 elements per thread) ----------------
__global__ __launch_bounds__(256) void cvt_bf16_k(const float* __restrict__ in,
                                                  bf16_t*      __restrict__ out, int n4) {
    int i = blockIdx.x * 256 + threadIdx.x;
    if (i < n4) {
        v4f v = ((const v4f*)in)[i];
        v4bf b;
        b[0] = (bf16_t)v[0];
        b[1] = (bf16_t)v[1];
        b[2] = (bf16_t)v[2];
        b[3] = (bf16_t)v[3];
        ((v4bf*)out)[i] = b;
    }
}

// ---------------- WMMA GEMM: out[n,o] = sum_k A[n,k] * W[o,k] ----------------
// A: [N_NODES, 256] bf16 (aggregated features), B = W: [256, 256] bf16 row-major.
// Block = 128 threads = 4 waves. Block owns a 16-row strip; wave w owns columns
// [w*64, w*64+64) as four 16x16 WMMA tiles. K loop: 8 steps of 16x16x32.
__global__ __launch_bounds__(128) void wmma_gemm_k(const bf16_t* __restrict__ A,
                                                   const bf16_t* __restrict__ B,
                                                   float*        __restrict__ out) {
    const int lane = threadIdx.x & 31;
    const int wave = threadIdx.x >> 5;      // 0..3
    const int row0 = blockIdx.x << 4;       // 16-row strip  (50000/16 = 3125 blocks)
    const int col0 = wave << 6;             // 64-column group
    const int l15  = lane & 15;
    const bool hi  = (lane & 16) != 0;

    // A fragment in this code's contiguous k order (not the ISA table order): lanes 0-15 -> row=lane, K 0..15;
    // lanes 16-31 -> row=lane-16, K 16..31
    const bf16_t* aPtr = A + (size_t)(row0 + l15) * DFEAT + (hi ? 16 : 0);
    // B fragment in the same contiguous k order as A (not the ISA table order): lanes 0-15 -> col=lane, K 0..15;
    // lanes 16-31 -> col=lane-16, K 16..31. B[k, col] = W[colGlobal, k] (row-major W)
    const bf16_t* bPtr = B + (size_t)(col0 + l15) * DFEAT + (hi ? 16 : 0);

    v8f acc0 = {}; v8f acc1 = {}; v8f acc2 = {}; v8f acc3 = {};

#pragma unroll
    for (int k0 = 0; k0 < DFEAT; k0 += 32) {
        v8bf alo = *(const v8bf*)(aPtr + k0);        // 16B load
        v8bf ahi = *(const v8bf*)(aPtr + k0 + 8);    // 16B load
        v16bf a = __builtin_shufflevector(alo, ahi, 0,1,2,3,4,5,6,7,8,9,10,11,12,13,14,15);

        {
            v8bf blo = *(const v8bf*)(bPtr + 0 * 16 * DFEAT + k0);
            v8bf bhi = *(const v8bf*)(bPtr + 0 * 16 * DFEAT + k0 + 8);
            v16bf b = __builtin_shufflevector(blo, bhi, 0,1,2,3,4,5,6,7,8,9,10,11,12,13,14,15);
            acc0 = __builtin_amdgcn_wmma_f32_16x16x32_bf16(false, a, false, b, (short)0, acc0, false, false);
        }
        {
            v8bf blo = *(const v8bf*)(bPtr + 1 * 16 * DFEAT + k0);
            v8bf bhi = *(const v8bf*)(bPtr + 1 * 16 * DFEAT + k0 + 8);
            v16bf b = __builtin_shufflevector(blo, bhi, 0,1,2,3,4,5,6,7,8,9,10,11,12,13,14,15);
            acc1 = __builtin_amdgcn_wmma_f32_16x16x32_bf16(false, a, false, b, (short)0, acc1, false, false);
        }
        {
            v8bf blo = *(const v8bf*)(bPtr + 2 * 16 * DFEAT + k0);
            v8bf bhi = *(const v8bf*)(bPtr + 2 * 16 * DFEAT + k0 + 8);
            v16bf b = __builtin_shufflevector(blo, bhi, 0,1,2,3,4,5,6,7,8,9,10,11,12,13,14,15);
            acc2 = __builtin_amdgcn_wmma_f32_16x16x32_bf16(false, a, false, b, (short)0, acc2, false, false);
        }
        {
            v8bf blo = *(const v8bf*)(bPtr + 3 * 16 * DFEAT + k0);
            v8bf bhi = *(const v8bf*)(bPtr + 3 * 16 * DFEAT + k0 + 8);
            v16bf b = __builtin_shufflevector(blo, bhi, 0,1,2,3,4,5,6,7,8,9,10,11,12,13,14,15);
            acc3 = __builtin_amdgcn_wmma_f32_16x16x32_bf16(false, a, false, b, (short)0, acc3, false, false);
        }
    }

    // C/D layout (32-bit 16x16): VGPR j: lanes 0-15 -> (M=j, N=lane); lanes 16-31 -> (M=j+8, N=lane-16)
    const int rbase = row0 + (hi ? 8 : 0);
#pragma unroll
    for (int j = 0; j < 8; ++j) {
        float* orow = out + (size_t)(rbase + j) * DFEAT + col0 + l15;
        orow[0]  = acc0[j];
        orow[16] = acc1[j];
        orow[32] = acc2[j];
        orow[48] = acc3[j];
    }
}

extern "C" void kernel_launch(void* const* d_in, const int* in_sizes, int n_in,
                              void* d_out, int out_size, void* d_ws, size_t ws_size,
                              hipStream_t stream) {
    const float* x  = (const float*)d_in[0];   // [50000, 256]
    const int*   ei = (const int*)d_in[1];     // [2, 300000]: row0 = dst, row1 = src
    const float* ew = (const float*)d_in[2];   // [300000]
    const float* W  = (const float*)d_in[3];   // [256, 256] (out, in)
    float* out = (float*)d_out;                // [50000, 256]

    // workspace layout
    char* ws = (char*)d_ws;
    float*  agg  = (float*)ws;                                  // 50000*256*4 = 51,200,000 B
    bf16_t* aggb = (bf16_t*)(ws + (size_t)51200000);            // 50000*256*2 = 25,600,000 B
    bf16_t* Wb   = (bf16_t*)(ws + (size_t)76800000);            // 256*256*2   =    131,072 B
    (void)in_sizes; (void)n_in; (void)out_size; (void)ws_size;

    const int agg_n4 = N_NODES * DFEAT / 4;    // 3,200,000 float4
    const int w_n4   = DFEAT * DFEAT / 4;      // 16,384 float4

    // 1) zero fp32 accumulator
    zero_f32_k<<<(agg_n4 + 255) / 256, 256, 0, stream>>>(agg, agg_n4);
    // 2) edge scatter with fp32 atomics: agg[dst] += w * x[src]
    scatter_k<<<(N_EDGES * 64 + 255) / 256, 256, 0, stream>>>(x, ei, ew, agg);
    // 3) convert agg and W to bf16 for the WMMA GEMM
    cvt_bf16_k<<<(agg_n4 + 255) / 256, 256, 0, stream>>>(agg, aggb, agg_n4);
    cvt_bf16_k<<<(w_n4 + 255) / 256, 256, 0, stream>>>(W, Wb, w_n4);
    // 4) out = agg @ W^T via v_wmma_f32_16x16x32_bf16 (writes every element of d_out)
    wmma_gemm_k<<<N_NODES / 16, 128, 0, stream>>>(aggb, Wb, out);
}
